// HeteroNeighborhoodAttention_80135499809235
// MI455X (gfx1250) — matrix operand order repaired and checked by reading, not compiled, not run on hardware
//
#include <hip/hip_runtime.h>
#include <hip/hip_bf16.h>

// ---------------------------------------------------------------------------
// MI455X (gfx1250) implementation: wave32 tiles of 16 edges/nodes driving
// v_wmma_f32_16x16x32_f16. Softmax max/normalization factored out so the edge
// pass is single-shot with f32 atomics for segment reduction.
// Round 2: batch-load all 8 B fragments per k-tile (memory-level parallelism
// inside the wave) and vectorize f16 LDS staging stores (b64 per float4).
// ---------------------------------------------------------------------------

typedef _Float16 v16h __attribute__((ext_vector_type(16)));
typedef _Float16 v8h  __attribute__((ext_vector_type(8)));
typedef _Float16 v4h  __attribute__((ext_vector_type(4)));
typedef float    v8f  __attribute__((ext_vector_type(8)));

#define WMMA_F16(a, b, c) \
    __builtin_amdgcn_wmma_f32_16x16x32_f16(false, (a), false, (b), (short)0, (c), false, false)

__device__ __forceinline__ v16h make16(v8h lo, v8h hi) {
    v16h r;
#pragma unroll
    for (int i = 0; i < 8; ++i) { r[i] = lo[i]; r[i + 8] = hi[i]; }
    return r;
}

__device__ __forceinline__ v4h cvt4(float4 v) {
    v4h h;
    h[0] = (_Float16)v.x; h[1] = (_Float16)v.y;
    h[2] = (_Float16)v.z; h[3] = (_Float16)v.w;
    return h;
}

// A fragment (16x32 f16, M=rows) from an LDS tile with row stride `ldk` halfs.
// Per ISA layout: lane half h, row m=lane&15 holds K = h*8+0..7 and 16+h*8+0..7
// -> two contiguous 16-byte chunks.
__device__ __forceinline__ v16h loadA_lds(const _Float16* X, int ldk, int kt, int lane) {
    const int half = (lane >> 4) & 1, m = lane & 15;
    const _Float16* p = X + m * ldk + kt * 32 + half * 8;
    v8h lo = *(const v8h*)(p);
    v8h hi = *(const v8h*)(p + 16);
    return make16(lo, hi);
}

// B fragment (32x16 f16) pre-packed lane-major: each lane reads 32 contiguous B.
__device__ __forceinline__ v16h loadB_glb(const _Float16* __restrict__ Wp, int kt, int nt, int lane) {
    const _Float16* p = Wp + ((size_t)((kt * 8 + nt) * 32 + lane) << 4);
    v8h lo = *(const v8h*)(p);
    v8h hi = *(const v8h*)(p + 8);
    return make16(lo, hi);
}

// acc[0..7] += X(16 x 32*NKT) @ W(32*NKT x 128).  All 8 B fragments of a
// k-tile are loaded before the wmma chain so the loads clause together and
// the waits retire progressively.
template <int NKT, int LDK>
__device__ __forceinline__ void gemm_acc(const _Float16* Xs,
                                         const _Float16* __restrict__ Wp,
                                         int lane, v8f acc[8]) {
#pragma unroll
    for (int kt = 0; kt < NKT; ++kt) {
        v16h a = loadA_lds(Xs, LDK, kt, lane);
        v16h b[8];
#pragma unroll
        for (int nt = 0; nt < 8; ++nt) b[nt] = loadB_glb(Wp, kt, nt, lane);
#pragma unroll
        for (int nt = 0; nt < 8; ++nt) acc[nt] = WMMA_F16(a, b[nt], acc[nt]);
    }
}

// residual MLP on a 16-row tile:
//   out = b2 + relu(X@W1 + b1) @ W2 + X@Wres       (C layout, 8 n-tiles of v8f)
template <int NKT, int LDK>
__device__ __forceinline__ void mlp16(const _Float16* Xs, _Float16* H1s,
                                      const _Float16* __restrict__ W1p,
                                      const _Float16* __restrict__ Wrp,
                                      const _Float16* __restrict__ W2p,
                                      const float* __restrict__ b1,
                                      const float* __restrict__ b2,
                                      int lane, v8f out[8]) {
    const int half = lane >> 4, sub = lane & 15;
    v8f acc[8];
    // ---- h1 = X @ W1 + b1
#pragma unroll
    for (int nt = 0; nt < 8; ++nt) {
        float b = b1[nt * 16 + sub];
#pragma unroll
        for (int r = 0; r < 8; ++r) acc[nt][r] = b;
    }
    gemm_acc<NKT, LDK>(Xs, W1p, lane, acc);
    // ---- stage relu(h1) to LDS in A-feed layout (row-major 16x128)
#pragma unroll
    for (int nt = 0; nt < 8; ++nt)
#pragma unroll
        for (int r = 0; r < 8; ++r)
            H1s[(r + 8 * half) * 128 + nt * 16 + sub] = (_Float16)fmaxf(acc[nt][r], 0.0f);
    // ---- acc = b2 + X @ Wres
#pragma unroll
    for (int nt = 0; nt < 8; ++nt) {
        float b = b2[nt * 16 + sub];
#pragma unroll
        for (int r = 0; r < 8; ++r) acc[nt][r] = b;
    }
    gemm_acc<NKT, LDK>(Xs, Wrp, lane, acc);
    // ---- acc += relu(h1) @ W2   (K = 128 -> 4 k-tiles)
    gemm_acc<4, 128>(H1s, W2p, lane, acc);
#pragma unroll
    for (int nt = 0; nt < 8; ++nt) out[nt] = acc[nt];
}

// ---------------------------------------------------------------------------
// Pack an fp32 [Krows x 128] weight into f16 B fragments (lane-major, K padded
// to KP with zeros).  out[((kt*8+nt)*32+lane)*16+e] = W[kt*32+(lane>>4)*8+(e&7)+(e>>3)*16][nt*16+(lane&15)]
// ---------------------------------------------------------------------------
__global__ void pack_w_kernel(const float* __restrict__ W, _Float16* __restrict__ out,
                              int Krows, int total) {
    int idx = blockIdx.x * blockDim.x + threadIdx.x;
    if (idx >= total) return;
    int e = idx & 15;
    int lane = (idx >> 4) & 31;
    int f = idx >> 9;
    int kt = f >> 3, nt = f & 7;
    int k = kt * 32 + ((lane >> 4) << 3) + (e & 7) + ((e >> 3) << 4);
    int n = nt * 16 + (lane & 15);
    float v = (k < Krows) ? W[(size_t)k * 128 + n] : 0.0f;
    out[idx] = (_Float16)v;
}

__global__ void zero_f32_kernel(float* __restrict__ p, size_t n) {
    size_t i = (size_t)blockIdx.x * blockDim.x + threadIdx.x;
    size_t stride = (size_t)gridDim.x * blockDim.x;
    for (; i < n; i += stride) p[i] = 0.0f;
}

// ---------------------------------------------------------------------------
// Edge pass: one wave per 16-edge tile.
//   x = concat(x_src[src], x_int[dst], attr)  (f16, K padded to KP)
//   k = resblock_k(x); v = resblock_v(x)
//   s[e,h] = (k[e, h*16:+16] . q[h]) / 4 ; ex = exp(s) (clamped)
//   denom[dst,h]  += ex        (atomic)
//   agg[dst, col] += v * ex    (atomic, unnormalized; normalized in node pass)
// ---------------------------------------------------------------------------
template <int DSRC, int DE, int KP>
__global__ __launch_bounds__(32) void edge_attn_kernel(
    const float* __restrict__ xsrc, const float* __restrict__ xint,
    const float* __restrict__ eattr, const int* __restrict__ src,
    const int* __restrict__ dst, int E,
    const _Float16* __restrict__ W1k, const _Float16* __restrict__ Wrk,
    const _Float16* __restrict__ W2k, const float* __restrict__ b1k,
    const float* __restrict__ b2k,
    const _Float16* __restrict__ W1v, const _Float16* __restrict__ Wrv,
    const _Float16* __restrict__ W2v, const float* __restrict__ b1v,
    const float* __restrict__ b2v,
    const float* __restrict__ q,
    float* __restrict__ agg, float* __restrict__ denom) {
    __shared__ __align__(16) _Float16 Xs[16 * KP];
    __shared__ __align__(16) _Float16 H1s[16 * 128];
    __shared__ float ExS[16 * 8];
    __shared__ int DstS[16];

    const int lane = threadIdx.x;
    const int half = lane >> 4, sub = lane & 15;
    const int e0 = blockIdx.x * 16;

    // ---- gather + f16 convert into LDS (E is a multiple of 16; clamp is a no-op)
    {
        int e = e0 + sub;
        if (e >= E) e = E - 1;
        if (half == 0) {
            int s = src[e];
            const float* r = xsrc + (size_t)s * DSRC;
#pragma unroll
            for (int j = 0; j < DSRC; j += 4)
                *(v4h*)&Xs[sub * KP + j] = cvt4(*(const float4*)(r + j));
        } else {
            int d = dst[e];
            DstS[sub] = d;
            const float* r = xint + (size_t)d * 64;
#pragma unroll
            for (int j = 0; j < 64; j += 4)
                *(v4h*)&Xs[sub * KP + DSRC + j] = cvt4(*(const float4*)(r + j));
            const float* a = eattr + (size_t)e * DE;
#pragma unroll
            for (int j = 0; j < DE; j += 4)
                *(v4h*)&Xs[sub * KP + DSRC + 64 + j] = cvt4(*(const float4*)(a + j));
#pragma unroll
            for (int j = DSRC + 64 + DE; j < KP; j += 4) {
                v4h z = {(_Float16)0.f, (_Float16)0.f, (_Float16)0.f, (_Float16)0.f};
                *(v4h*)&Xs[sub * KP + j] = z;
            }
        }
    }
    __syncthreads();

    constexpr int NKT = KP / 32;

    // ---- k MLP + scores
    v8f kacc[8];
    mlp16<NKT, KP>(Xs, H1s, W1k, Wrk, W2k, b1k, b2k, lane, kacc);
#pragma unroll
    for (int nt = 0; nt < 8; ++nt) {           // n-tile == attention head
        float qc = q[nt * 16 + sub];
#pragma unroll
        for (int r = 0; r < 8; ++r) {
            float x = kacc[nt][r] * qc;
            x += __shfl_xor(x, 1, 32);
            x += __shfl_xor(x, 2, 32);
            x += __shfl_xor(x, 4, 32);
            x += __shfl_xor(x, 8, 32);         // sum over 16 cols of this head
            float ex = __expf(fminf(x * 0.25f, 30.0f));  // softmax max-shift cancels
            if (sub == 0) {
                int M = r + 8 * half;
                ExS[M * 8 + nt] = ex;
                atomicAdd(&denom[(size_t)DstS[M] * 8 + nt], ex);
            }
        }
    }
    __syncthreads();

    // ---- v MLP + unnormalized scatter
    v8f vacc[8];
    mlp16<NKT, KP>(Xs, H1s, W1v, Wrv, W2v, b1v, b2v, lane, vacc);
#pragma unroll
    for (int nt = 0; nt < 8; ++nt)
#pragma unroll
        for (int r = 0; r < 8; ++r) {
            int M = r + 8 * half;
            float ex = ExS[M * 8 + nt];
            atomicAdd(&agg[(size_t)DstS[M] * 128 + nt * 16 + sub], vacc[nt][r] * ex);
        }
}

// ---------------------------------------------------------------------------
// Node pass: normalize aggregates per head, run the 256->128 residual block,
// add x_int @ Wid, final relu.
// ---------------------------------------------------------------------------
__global__ __launch_bounds__(32) void node_update_kernel(
    const float* __restrict__ agg1, const float* __restrict__ den1,
    const float* __restrict__ agg2, const float* __restrict__ den2,
    const float* __restrict__ xint, int N,
    const _Float16* __restrict__ W1u, const _Float16* __restrict__ Wru,
    const _Float16* __restrict__ W2u, const float* __restrict__ b1u,
    const float* __restrict__ b2u,
    const _Float16* __restrict__ Widp, float* __restrict__ out) {
    __shared__ __align__(16) _Float16 Xs[16 * 256];
    __shared__ __align__(16) _Float16 XI[16 * 64];
    __shared__ __align__(16) _Float16 H1s[16 * 128];

    const int lane = threadIdx.x;
    const int half = lane >> 4, sub = lane & 15;
    const int n0 = blockIdx.x * 16;

    {
        int node = n0 + sub;
        if (node >= N) node = N - 1;
        const float* ag = (half == 0) ? agg1 : agg2;
        const float* dn = (half == 0) ? den1 : den2;
        const float* a = ag + (size_t)node * 128;
        const float* d = dn + (size_t)node * 8;
        float dl[8];
#pragma unroll
        for (int h = 0; h < 8; ++h) dl[h] = 1.0f / (d[h] + 1e-16f);
#pragma unroll
        for (int c = 0; c < 128; c += 4) {
            float4 v = *(const float4*)(a + c);
            float s = dl[c >> 4];
            v.x *= s; v.y *= s; v.z *= s; v.w *= s;
            *(v4h*)&Xs[sub * 256 + half * 128 + c] = cvt4(v);
        }
        if (half == 0) {
            const float* xr = xint + (size_t)node * 64;
#pragma unroll
            for (int c = 0; c < 64; c += 4)
                *(v4h*)&XI[sub * 64 + c] = cvt4(*(const float4*)(xr + c));
        }
    }
    __syncthreads();

    v8f acc[8];
    mlp16<8, 256>(Xs, H1s, W1u, Wru, W2u, b1u, b2u, lane, acc);
    // + x_int @ Wid  (K = 64 -> 2 k-tiles)
    gemm_acc<2, 64>(XI, Widp, lane, acc);
#pragma unroll
    for (int nt = 0; nt < 8; ++nt)
#pragma unroll
        for (int r = 0; r < 8; ++r) {
            int node = n0 + r + 8 * half;
            if (node < N)
                out[(size_t)node * 128 + nt * 16 + sub] = fmaxf(acc[nt][r], 0.0f);
        }
}

// ---------------------------------------------------------------------------
// Host launcher
// ---------------------------------------------------------------------------
extern "C" void kernel_launch(void* const* d_in, const int* in_sizes, int n_in,
                              void* d_out, int out_size, void* d_ws, size_t ws_size,
                              hipStream_t stream) {
    (void)n_in; (void)out_size; (void)ws_size;
    // inputs (nested dicts flatten with sorted keys: W1,W2,Wres,b1,b2 / k,q,v)
    const float* x_lane = (const float*)d_in[0];
    const float* x_mov  = (const float*)d_in[1];
    const float* x_int  = (const float*)d_in[2];
    const float* ea1    = (const float*)d_in[3];
    const float* ea2    = (const float*)d_in[4];
    const int*   src1   = (const int*)d_in[5];
    const int*   dst1   = (const int*)d_in[6];
    const int*   src2   = (const int*)d_in[7];
    const int*   dst2   = (const int*)d_in[8];
    const float* p1kW1 = (const float*)d_in[9];
    const float* p1kW2 = (const float*)d_in[10];
    const float* p1kWr = (const float*)d_in[11];
    const float* p1kb1 = (const float*)d_in[12];
    const float* p1kb2 = (const float*)d_in[13];
    const float* p1q   = (const float*)d_in[14];
    const float* p1vW1 = (const float*)d_in[15];
    const float* p1vW2 = (const float*)d_in[16];
    const float* p1vWr = (const float*)d_in[17];
    const float* p1vb1 = (const float*)d_in[18];
    const float* p1vb2 = (const float*)d_in[19];
    const float* p2kW1 = (const float*)d_in[20];
    const float* p2kW2 = (const float*)d_in[21];
    const float* p2kWr = (const float*)d_in[22];
    const float* p2kb1 = (const float*)d_in[23];
    const float* p2kb2 = (const float*)d_in[24];
    const float* p2q   = (const float*)d_in[25];
    const float* p2vW1 = (const float*)d_in[26];
    const float* p2vW2 = (const float*)d_in[27];
    const float* p2vWr = (const float*)d_in[28];
    const float* p2vb1 = (const float*)d_in[29];
    const float* p2vb2 = (const float*)d_in[30];
    const float* puW1  = (const float*)d_in[31];
    const float* puW2  = (const float*)d_in[32];
    const float* puWr  = (const float*)d_in[33];
    const float* pub1  = (const float*)d_in[34];
    const float* pub2  = (const float*)d_in[35];
    const float* Wid   = (const float*)d_in[36];

    const int E1 = in_sizes[5];
    const int E2 = in_sizes[7];
    const int N  = in_sizes[2] / 64;

    // ---- workspace carve-out (256B-aligned cursor)
    char* ws = (char*)d_ws;
    auto alloc = [&](size_t bytes) -> char* {
        char* p = ws;
        ws += (bytes + 255) & ~(size_t)255;
        return p;
    };
    const int FR160 = 5 * 8 * 512, FR128 = 4 * 8 * 512, FR256 = 8 * 8 * 512, FR64 = 2 * 8 * 512;
    _Float16* p1kW1p = (_Float16*)alloc(FR160 * 2);
    _Float16* p1kWrp = (_Float16*)alloc(FR160 * 2);
    _Float16* p1kW2p = (_Float16*)alloc(FR128 * 2);
    _Float16* p1vW1p = (_Float16*)alloc(FR160 * 2);
    _Float16* p1vWrp = (_Float16*)alloc(FR160 * 2);
    _Float16* p1vW2p = (_Float16*)alloc(FR128 * 2);
    _Float16* p2kW1p = (_Float16*)alloc(FR128 * 2);
    _Float16* p2kWrp = (_Float16*)alloc(FR128 * 2);
    _Float16* p2kW2p = (_Float16*)alloc(FR128 * 2);
    _Float16* p2vW1p = (_Float16*)alloc(FR128 * 2);
    _Float16* p2vWrp = (_Float16*)alloc(FR128 * 2);
    _Float16* p2vW2p = (_Float16*)alloc(FR128 * 2);
    _Float16* puW1p  = (_Float16*)alloc(FR256 * 2);
    _Float16* puWrp  = (_Float16*)alloc(FR256 * 2);
    _Float16* puW2p  = (_Float16*)alloc(FR128 * 2);
    _Float16* Widp   = (_Float16*)alloc(FR64 * 2);
    float* agg1 = (float*)alloc((size_t)N * 128 * 4);
    float* agg2 = (float*)alloc((size_t)N * 128 * 4);
    float* den1 = (float*)alloc((size_t)N * 8 * 4);
    float* den2 = (float*)alloc((size_t)N * 8 * 4);

    // ---- zero segment-reduction buffers (contiguous: agg1,agg2,den1,den2)
    zero_f32_kernel<<<1024, 256, 0, stream>>>(agg1, (size_t)N * 128 * 2 + (size_t)N * 8 * 2);

    // ---- pack weights into f16 B fragments
    auto pack = [&](const float* W, _Float16* out, int Krows, int total) {
        pack_w_kernel<<<(total + 255) / 256, 256, 0, stream>>>(W, out, Krows, total);
    };
    pack(p1kW1, p1kW1p, 144, FR160); pack(p1kWr, p1kWrp, 144, FR160); pack(p1kW2, p1kW2p, 128, FR128);
    pack(p1vW1, p1vW1p, 144, FR160); pack(p1vWr, p1vWrp, 144, FR160); pack(p1vW2, p1vW2p, 128, FR128);
    pack(p2kW1, p2kW1p, 104, FR128); pack(p2kWr, p2kWrp, 104, FR128); pack(p2kW2, p2kW2p, 128, FR128);
    pack(p2vW1, p2vW1p, 104, FR128); pack(p2vWr, p2vWrp, 104, FR128); pack(p2vW2, p2vW2p, 128, FR128);
    pack(puW1,  puW1p,  256, FR256); pack(puWr,  puWrp,  256, FR256); pack(puW2,  puW2p,  128, FR128);
    pack(Wid,   Widp,   64,  FR64);

    // ---- edge passes (one wave per 16-edge tile)
    edge_attn_kernel<64, 16, 160><<<(E1 + 15) / 16, 32, 0, stream>>>(
        x_lane, x_int, ea1, src1, dst1, E1,
        p1kW1p, p1kWrp, p1kW2p, p1kb1, p1kb2,
        p1vW1p, p1vWrp, p1vW2p, p1vb1, p1vb2,
        p1q, agg1, den1);
    edge_attn_kernel<32, 8, 128><<<(E2 + 15) / 16, 32, 0, stream>>>(
        x_mov, x_int, ea2, src2, dst2, E2,
        p2kW1p, p2kWrp, p2kW2p, p2kb1, p2kb2,
        p2vW1p, p2vWrp, p2vW2p, p2vb1, p2vb2,
        p2q, agg2, den2);

    // ---- node update
    node_update_kernel<<<(N + 15) / 16, 32, 0, stream>>>(
        agg1, den1, agg2, den2, x_int, N,
        puW1p, puWrp, puW2p, pub1, pub2, Widp, (float*)d_out);
}
